// ATNLPmodel_17892833755479
// MI455X (gfx1250) — matrix operand order repaired and checked by reading, not compiled, not run on hardware
//
#include <hip/hip_runtime.h>
#include <hip/hip_bf16.h>
#include <float.h>
#include <math.h>

typedef __attribute__((ext_vector_type(16))) _Float16 v16h;
typedef __attribute__((ext_vector_type(8)))  float    v8f;
typedef __attribute__((ext_vector_type(4)))  unsigned int v4u;
typedef __attribute__((ext_vector_type(8)))  int      v8i;
typedef __attribute__((ext_vector_type(4)))  int      v4i;

#define B_      8
#define LB      128
#define LC      512
#define L2_     128
#define R_      4
#define P_      2
#define NDB     16384
#define KDIM    2048      // 16*128 flattened
#define MTILES  10        // 160 query rows / 16
#define KSTEPS  64        // 2048 / 32
#define NTILES  1024      // 16384 / 16
#define STAGE   4         // K-steps staged per TDM transfer (40 KB LDS)

#if __has_builtin(__builtin_amdgcn_tensor_load_to_lds) && __has_builtin(__builtin_amdgcn_s_wait_tensorcnt)
#define USE_TDM 1
#else
#define USE_TDM 0
#endif

// ---------------------------------------------------------------------------
// Kernel 1: per-character token ids (B x Lc), searching the offset table.
// ---------------------------------------------------------------------------
__global__ void k_ids(const long long* __restrict__ ids64,
                      const long long* __restrict__ offs,
                      int* __restrict__ ids) {
  int t = blockIdx.x * blockDim.x + threadIdx.x;
  if (t >= B_ * LC) return;
  int b = t / LC, c = t % LC;
  int id = 0;  // PAD_ID
  for (int tok = 0; tok < LB; ++tok) {
    long long s = offs[((long long)b * LB + tok) * 2 + 0];
    long long e = offs[((long long)b * LB + tok) * 2 + 1];
    if ((long long)c >= s && (long long)c < e) { id = (int)ids64[b * LB + tok]; break; }
  }
  ids[t] = id;
}

// ---------------------------------------------------------------------------
// Kernel 2: per-unit inverse Frobenius norm of the binary query (popcount).
// ---------------------------------------------------------------------------
__global__ void k_invq(const int* __restrict__ ids, const int* __restrict__ swi,
                       float* __restrict__ invq) {
  int u = threadIdx.x;
  if (u >= 32) return;
  int b = u >> 2, r = u & 3;
  int st = swi[0] + r * ((LC - L2_) / (R_ - 1));
  st = st < 0 ? 0 : (st > LC - L2_ ? LC - L2_ : st);
  int sum = 0;
  for (int j = 0; j < L2_; ++j)
    sum += __popc(ids[b * LC + st + j] & 0xFFFF);
  invq[u] = 1.0f / (sqrtf((float)sum) + 1e-8f);
}

// ---------------------------------------------------------------------------
// Kernel 3: build 160x2048 f16 query matrix (5 shifts x 32 units) directly in
// the documented WMMA 16-bit A-matrix layout:
//   buffer[((mt*64+ks)*32 + lane)*16 + e], kk = (e&7) + 16*(e>>3) + 8*(lane>>4)
// ---------------------------------------------------------------------------
__global__ void k_buildQ(const int* __restrict__ ids, const int* __restrict__ swi,
                         _Float16* __restrict__ Q) {
  int idx = blockIdx.x * blockDim.x + threadIdx.x;
  if (idx >= MTILES * KSTEPS * 32 * 16) return;
  int e    = idx & 15;
  int lane = (idx >> 4) & 31;
  int ks   = (idx >> 9) & 63;
  int mt   = idx >> 15;
  int m  = lane & 15, hi = lane >> 4;
  int kk = (e & 7) + ((e >> 3) << 4) + (hi << 3);
  int k  = ks * 32 + kk;
  int mg = mt * 16 + m;        // global query row 0..159
  int sidx = mg >> 5;          // shift index 0..4
  int u    = mg & 31;          // unit 0..31
  int s = sidx - P_;           // shift -2..2
  int b = u >> 2, r = u & 3;
  int ch = k >> 7, j = k & 127;
  int jsrc = (j - s + 128) & 127;   // jnp.roll semantics
  int st = swi[0] + r * ((LC - L2_) / (R_ - 1));
  st = st < 0 ? 0 : (st > LC - L2_ ? LC - L2_ : st);
  int id = ids[b * LC + st + jsrc];
  Q[idx] = (_Float16)(float)((id >> ch) & 1);
}

// ---------------------------------------------------------------------------
// Kernel 4: fused GEMM + DB normalization + shift-max + per-tile reductions.
// 1 wave = 16 DB rows x all 160 query rows, K=2048 in 64 steps of 32.
// A staged STAGE K-steps at a time via the Tensor Data Mover (2-D tile:
// dim0 = STAGE*256 dwords contiguous, dim1 = 10 M-tiles, stride 16384 dwords),
// published to the 8 waves with s_wait_tensorcnt + barrier. B converted
// f32->f16 inline; ||db_row||^2 accumulated by the same lanes that feed WMMA-B.
// ---------------------------------------------------------------------------
__global__ void __launch_bounds__(256)
k_gemm(const float* __restrict__ db, const _Float16* __restrict__ Q,
       const float* __restrict__ invq,
       float* __restrict__ Pmax, int* __restrict__ Pidx, float* __restrict__ Psum) {
  __shared__ __align__(32) _Float16 Alds[MTILES * 512 * STAGE];   // 40 KB
  const int tid   = threadIdx.x;
  const int lane  = tid & 31;
  const int wid   = tid >> 5;
  const int gw    = blockIdx.x * 8 + wid;   // global wave / N-tile id, 0..1023
  const int nbase = gw * 16;
  const int n     = lane & 15;              // B column = db row within tile
  const int hi    = lane >> 4;              // K-half select

  const float* rowp = db + (size_t)(nbase + n) * KDIM + hi * 8;

  v8f acc[MTILES];
  v8f zero = {0.f, 0.f, 0.f, 0.f, 0.f, 0.f, 0.f, 0.f};
#pragma unroll
  for (int i = 0; i < MTILES; ++i) acc[i] = zero;
  float ss = 0.f;

#if !USE_TDM
  const uint32_t* Qd = (const uint32_t*)Q;
  uint32_t* Ad = (uint32_t*)Alds;
#endif

  for (int ks0 = 0; ks0 < KSTEPS; ks0 += STAGE) {
    __syncthreads();                       // previous stage fully consumed
#if USE_TDM
    if (wid == 0) {
      // ---- Tensor DMA descriptor (ISA 8.3/8.4), 2-D tile ----
      unsigned long long ga =
          (unsigned long long)(uintptr_t)Q + (unsigned long long)ks0 * 1024ull;
      unsigned int lds = (unsigned int)(uintptr_t)&Alds[0];  // low 32b = LDS offset
      v4u g0;
      g0[0] = 1u;                                   // count=1, user descriptor
      g0[1] = lds;                                  // lds_addr
      g0[2] = (unsigned int)(ga & 0xFFFFFFFFull);   // global_addr[31:0]
      g0[3] = (unsigned int)((ga >> 32) & 0x01FFFFFFu) | (2u << 30); // [56:32]|type=2
      v8i g1;
      g1[0] = 0x00020000;                           // data_size=4B, no multicast
      g1[1] = (int)((STAGE * 256) & 0xFFFF) << 16;  // tensor_dim0 lo16 (=1024 dwords)
      g1[2] = (int)(((STAGE * 256) >> 16) & 0xFFFF) | (MTILES << 16); // dim0 hi | dim1 lo
      g1[3] = (int)((STAGE * 256) << 16);           // dim1 hi(0) | tile_dim0=1024
      g1[4] = MTILES;                               // tile_dim1=10, tile_dim2=0
      g1[5] = KSTEPS * 256;                         // tensor_dim0_stride = 16384 dwords
      g1[6] = 0;                                    // stride0 hi | stride1 lo
      g1[7] = 0;                                    // stride1 hi
      v4i z4 = {0, 0, 0, 0};                        // groups 2/3 unused (2-D)
      v8i z8 = {0, 0, 0, 0, 0, 0, 0, 0};
      __builtin_amdgcn_tensor_load_to_lds(g0, g1, z4, z4, z8, 0);
      __builtin_amdgcn_s_wait_tensorcnt(0);
    }
#else
    // Fallback: cooperative copy (global -> VGPR -> LDS)
#pragma unroll
    for (int mt = 0; mt < MTILES; ++mt)
      for (int w = tid; w < 256 * STAGE; w += 256)
        Ad[mt * 256 * STAGE + w] = Qd[(mt * 64 + ks0) * 256 + w];
#endif
    __syncthreads();                       // publish stage to all 8 waves

#pragma unroll
    for (int sks = 0; sks < STAGE; ++sks) {
      const int ks = ks0 + sks;
      const float* p = rowp + ks * 32;
      if (ks + 1 < KSTEPS) __builtin_prefetch(p + 32, 0, 1);  // global_prefetch_b8
      float4 f0 = *(const float4*)(p + 0);
      float4 f1 = *(const float4*)(p + 4);
      float4 f2 = *(const float4*)(p + 16);
      float4 f3 = *(const float4*)(p + 20);
      ss += f0.x*f0.x + f0.y*f0.y + f0.z*f0.z + f0.w*f0.w
          + f1.x*f1.x + f1.y*f1.y + f1.z*f1.z + f1.w*f1.w
          + f2.x*f2.x + f2.y*f2.y + f2.z*f2.z + f2.w*f2.w
          + f3.x*f3.x + f3.y*f3.y + f3.z*f3.z + f3.w*f3.w;
      v16h bv;
      bv[0]=(_Float16)f0.x;  bv[1]=(_Float16)f0.y;  bv[2]=(_Float16)f0.z;  bv[3]=(_Float16)f0.w;
      bv[4]=(_Float16)f1.x;  bv[5]=(_Float16)f1.y;  bv[6]=(_Float16)f1.z;  bv[7]=(_Float16)f1.w;
      bv[8]=(_Float16)f2.x;  bv[9]=(_Float16)f2.y;  bv[10]=(_Float16)f2.z; bv[11]=(_Float16)f2.w;
      bv[12]=(_Float16)f3.x; bv[13]=(_Float16)f3.y; bv[14]=(_Float16)f3.z; bv[15]=(_Float16)f3.w;

#pragma unroll
      for (int mt = 0; mt < MTILES; ++mt) {
        v16h a = *(const v16h*)(&Alds[(mt * STAGE + sks) * 512 + lane * 16]);
        acc[mt] = __builtin_amdgcn_wmma_f32_16x16x32_f16(
            false, a, false, bv, (short)0, acc[mt], false, false);
      }
    }
  }

  // lanes n and n+16 together saw the whole db row -> combine sumsq
  float ss2 = ss + __shfl_xor(ss, 16, 32);
  float invdb = 1.0f / (sqrtf(ss2) + 1e-8f);

  // D layout: VGPR v, lane -> row m = v + 8*hi, col = n. Even tiles = unit m,
  // odd tiles = unit 16+m, across the 5 shifts -> max over shifts in registers.
#pragma unroll
  for (int v = 0; v < 8; ++v) {
    int mrow = v + (hi << 3);     // 0..15
    float iq0 = invq[mrow];
    float iq1 = invq[16 + mrow];
    float m0 = acc[0][v];
    m0 = fmaxf(m0, acc[2][v]); m0 = fmaxf(m0, acc[4][v]);
    m0 = fmaxf(m0, acc[6][v]); m0 = fmaxf(m0, acc[8][v]);
    float m1 = acc[1][v];
    m1 = fmaxf(m1, acc[3][v]); m1 = fmaxf(m1, acc[5][v]);
    m1 = fmaxf(m1, acc[7][v]); m1 = fmaxf(m1, acc[9][v]);
    float s0 = m0 * iq0 * invdb;
    float s1 = m1 * iq1 * invdb;

    float v0 = s0, sum0 = s0; int i0 = n;
    float v1 = s1, sum1 = s1; int i1 = n;
#pragma unroll
    for (int d = 1; d < 16; d <<= 1) {     // stays within 16-lane half
      float ov = __shfl_xor(v0, d, 32); int oi = __shfl_xor(i0, d, 32);
      if (ov > v0 || (ov == v0 && oi < i0)) { v0 = ov; i0 = oi; }
      sum0 += __shfl_xor(sum0, d, 32);
      float ow = __shfl_xor(v1, d, 32); int oj = __shfl_xor(i1, d, 32);
      if (ow > v1 || (ow == v1 && oj < i1)) { v1 = ow; i1 = oj; }
      sum1 += __shfl_xor(sum1, d, 32);
    }
    if (n == 0) {
      int u0 = mrow, u1 = 16 + mrow;
      Pmax[u0 * NTILES + gw] = v0;  Pidx[u0 * NTILES + gw] = nbase + i0;
      Psum[u0 * NTILES + gw] = sum0;
      Pmax[u1 * NTILES + gw] = v1;  Pidx[u1 * NTILES + gw] = nbase + i1;
      Psum[u1 * NTILES + gw] = sum1;
    }
  }
}

// ---------------------------------------------------------------------------
// Kernel 5: final reduce per unit (max / argmax -> class / mean).
// ---------------------------------------------------------------------------
__global__ void k_reduce(const float* __restrict__ Pmax, const int* __restrict__ Pidx,
                         const float* __restrict__ Psum, const int* __restrict__ db_classes,
                         float* __restrict__ out) {
  int u = blockIdx.x;
  int tid = threadIdx.x;
  float vm = -FLT_MAX; int vi = 0x7fffffff; float su = 0.f;
  for (int t = tid; t < NTILES; t += blockDim.x) {
    float x = Pmax[u * NTILES + t];
    int  ix = Pidx[u * NTILES + t];
    if (x > vm || (x == vm && ix < vi)) { vm = x; vi = ix; }
    su += Psum[u * NTILES + t];
  }
  __shared__ float smax[256]; __shared__ int sidx[256]; __shared__ float ssum[256];
  smax[tid] = vm; sidx[tid] = vi; ssum[tid] = su;
  __syncthreads();
  for (int step = 128; step > 0; step >>= 1) {
    if (tid < step) {
      float x = smax[tid + step]; int ix = sidx[tid + step];
      if (x > smax[tid] || (x == smax[tid] && ix < sidx[tid])) { smax[tid] = x; sidx[tid] = ix; }
      ssum[tid] += ssum[tid + step];
    }
    __syncthreads();
  }
  if (tid == 0) {
    out[u]      = smax[0];
    out[32 + u] = (float)db_classes[sidx[0]];
    out[64 + u] = ssum[0] / (float)NDB;
  }
}

// ---------------------------------------------------------------------------
extern "C" void kernel_launch(void* const* d_in, const int* in_sizes, int n_in,
                              void* d_out, int out_size, void* d_ws, size_t ws_size,
                              hipStream_t stream) {
  (void)in_sizes; (void)n_in; (void)out_size; (void)ws_size;
  const long long* ids64      = (const long long*)d_in[0];  // bert_input_ids (int64)
  const long long* offs       = (const long long*)d_in[1];  // bert_offsets   (int64)
  const float*     db         = (const float*)d_in[2];      // database
  const int*       db_classes = (const int*)d_in[3];
  const int*       swi        = (const int*)d_in[4];        // slidingWindowIndex

  char* ws = (char*)d_ws;
  int*      ids  = (int*)ws;                      //   0 .. 16 KB   (8*512 ids)
  float*    invq = (float*)(ws + 16384);          //  16 KB         (32 floats)
  _Float16* Q    = (_Float16*)(ws + 32768);       //  32 KB .. 672 KB (160*2048 f16)
  float*    Pmax = (float*)(ws + 720896);         // 704 KB (32*1024 f32)
  int*      Pidx = (int*)  (ws + 851968);         // 832 KB (32*1024 i32)
  float*    Psum = (float*)(ws + 983040);         // 960 KB (32*1024 f32)

  k_ids   <<<16,   256, 0, stream>>>(ids64, offs, ids);
  k_invq  <<<1,     32, 0, stream>>>(ids, swi, invq);
  k_buildQ<<<1280, 256, 0, stream>>>(ids, swi, Q);
  k_gemm  <<<128,  256, 0, stream>>>(db, Q, invq, Pmax, Pidx, Psum);
  k_reduce<<<32,   256, 0, stream>>>(Pmax, Pidx, Psum, db_classes, (float*)d_out);
}
